// Set2Set_64476049047870
// MI455X (gfx1250) — matrix operand order repaired and checked by reading, not compiled, not run on hardware
//
#include <hip/hip_runtime.h>
#include <hip/hip_bf16.h>
#include <math.h>

typedef __attribute__((ext_vector_type(16))) _Float16 v16h;
typedef __attribute__((ext_vector_type(8)))  _Float16 v8h;
typedef __attribute__((ext_vector_type(8)))  float    v8f;

#define FDIM  256          // F
#define KDIM  768          // 2F (q_star) + F (h)
#define NGATE 1024         // 4F
#define TSTEPS 6

// ---------------- prep kernels ----------------

// Fuse [Wih | Whh] into one f16 row-major [1024 x 768] matrix.
// Row n of this matrix is exactly the contiguous-K B-fragment lane n needs.
__global__ void prep_weights(const float* __restrict__ Wih,
                             const float* __restrict__ Whh,
                             _Float16* __restrict__ Wf16) {
    int idx = blockIdx.x * blockDim.x + threadIdx.x;   // over 1024*768
    if (idx >= NGATE * KDIM) return;
    int r = idx / KDIM, k = idx % KDIM;
    float v = (k < 2 * FDIM) ? Wih[(size_t)r * (2 * FDIM) + k]
                             : Whh[(size_t)r * FDIM + (k - 2 * FDIM)];
    Wf16[idx] = (_Float16)v;
}

__global__ void prep_bias(const float* __restrict__ bih,
                          const float* __restrict__ bhh,
                          float* __restrict__ bsum) {
    int j = blockIdx.x * blockDim.x + threadIdx.x;
    if (j < NGATE) bsum[j] = bih[j] + bhh[j];
}

// batch[] is sorted: compute per-graph start offsets (start[G] = N).
__global__ void seg_start(const int* __restrict__ batch, int* __restrict__ start,
                          int N, int G) {
    int n = blockIdx.x * blockDim.x + threadIdx.x;
    if (n >= N) return;
    int b  = batch[n];
    int pb = (n == 0) ? -1 : batch[n - 1];
    for (int g = pb + 1; g <= b; ++g) start[g] = n;
    if (n == N - 1)
        for (int g = b + 1; g <= G; ++g) start[g] = N;
}

// ---------------- LSTM gate GEMM: gates[G,1024] = Z[G,768] * Wf16[1024,768]^T ----------------
// One wave per 16x16 output tile, K loop of 24 x v_wmma_f32_16x16x32_f16.
__global__ void __launch_bounds__(256) gemm_gates(const _Float16* __restrict__ Z,
                                                  const _Float16* __restrict__ W,
                                                  float* __restrict__ gates,
                                                  int G) {
    int wave = (blockIdx.x * blockDim.x + threadIdx.x) >> 5;
    int lane = threadIdx.x & 31;
    const int tiles_n = NGATE / 16;                 // 64
    int tm = wave / tiles_n, tn = wave % tiles_n;
    int row_base = tm * 16, col_base = tn * 16;
    if (row_base >= G) return;
    int hs = lane >> 4;                             // half-wave select
    int lr = lane & 15;

    // A fragment (16-bit A 16x32): lane group 0 holds K[0..15];
    // group 1 holds K[16..31] of its row.
    const _Float16* arow = Z + (size_t)(row_base + lr) * KDIM;
    // B fragment (16-bit B 32x16): lane = column, 16 contiguous K values.
    const _Float16* brow = W + (size_t)(col_base + lr) * KDIM;

    v8f acc = {};
#pragma unroll
    for (int kk = 0; kk < KDIM; kk += 32) {
        v16h a = *(const v16h*)(arow + kk + hs * 16);
        v16h b = *(const v16h*)(brow + kk + hs * 16);
        acc = __builtin_amdgcn_wmma_f32_16x16x32_f16(
                  false, a, false, b, (short)0, acc, false, false);
    }
    // D layout: element r -> D[row_base + hs*8 + r][col_base + lr]
    float* out = gates + (size_t)(row_base + hs * 8) * NGATE + col_base + lr;
#pragma unroll
    for (int r = 0; r < 8; ++r)
        out[(size_t)r * NGATE] = acc[r];
}

// ---------------- LSTM pointwise: activations + state update ----------------
__device__ __forceinline__ float sigmf(float v) { return 1.0f / (1.0f + __expf(-v)); }

__global__ void lstm_pointwise(const float* __restrict__ gates,
                               const float* __restrict__ bsum,
                               float* __restrict__ h, float* __restrict__ c,
                               float* __restrict__ q_star, _Float16* __restrict__ Z,
                               int G) {
    int idx = blockIdx.x * blockDim.x + threadIdx.x;
    if (idx >= G * FDIM) return;
    int g = idx / FDIM, j = idx % FDIM;
    const float* gr = gates + (size_t)g * NGATE;
    float gi = gr[j]            + bsum[j];
    float gf = gr[j + FDIM]     + bsum[j + FDIM];
    float gg = gr[j + 2 * FDIM] + bsum[j + 2 * FDIM];
    float go = gr[j + 3 * FDIM] + bsum[j + 3 * FDIM];
    float cn = sigmf(gf) * c[idx] + sigmf(gi) * tanhf(gg);
    float hn = sigmf(go) * tanhf(cn);
    c[idx] = cn;
    h[idx] = hn;
    q_star[(size_t)g * (2 * FDIM) + j] = hn;       // q half of output
    _Float16 hf = (_Float16)hn;
    Z[(size_t)g * KDIM + j] = hf;                  // q part of next GEMM input
    Z[(size_t)g * KDIM + 2 * FDIM + j] = hf;       // h part of next GEMM input
}

// ---------------- fused segment softmax-attention readout ----------------
// One 256-thread workgroup (8 waves) per graph; batch sorted -> contiguous segment.
__global__ void __launch_bounds__(256) attn_fused(const float* __restrict__ x,
                                                  const float* __restrict__ h,
                                                  const int* __restrict__ start,
                                                  float* __restrict__ e_buf,
                                                  float* __restrict__ q_star,
                                                  _Float16* __restrict__ Z) {
    int g   = blockIdx.x;
    int s   = start[g], e = start[g + 1];
    int tid = threadIdx.x, wave = tid >> 5, lane = tid & 31;

    __shared__ float qs[FDIM];
    __shared__ float red[8];
    __shared__ float sred[256];
    __shared__ float racc[FDIM];

    qs[tid]   = h[(size_t)g * FDIM + tid];
    racc[tid] = 0.0f;
    __syncthreads();

    float4 q0 = *(const float4*)(qs + lane * 8);
    float4 q1 = *(const float4*)(qs + lane * 8 + 4);

    // pass 1: e[n] = <x[n], q>, track running max (wave per node, 8 ch/lane)
    float wmax = -INFINITY;
    for (int n = s + wave; n < e; n += 8) {
        const float4* xp = (const float4*)(x + (size_t)n * FDIM + lane * 8);
        float4 x0 = xp[0], x1 = xp[1];
        float d = x0.x * q0.x + x0.y * q0.y + x0.z * q0.z + x0.w * q0.w
                + x1.x * q1.x + x1.y * q1.y + x1.z * q1.z + x1.w * q1.w;
#pragma unroll
        for (int off = 16; off > 0; off >>= 1) d += __shfl_xor(d, off, 32);
        if (lane == 0) e_buf[n] = d;
        wmax = fmaxf(wmax, d);
    }
    if (lane == 0) red[wave] = wmax;
    __syncthreads();
    float m = -INFINITY;
#pragma unroll
    for (int w = 0; w < 8; ++w) m = fmaxf(m, red[w]);

    // pass 2: denom = sum exp(e - m)
    float wsum = 0.0f;
    for (int n = s + tid; n < e; n += 256) wsum += __expf(e_buf[n] - m);
    sred[tid] = wsum;
    __syncthreads();
    for (int st = 128; st > 0; st >>= 1) {
        if (tid < st) sred[tid] += sred[tid + st];
        __syncthreads();
    }
    float inv = 1.0f / (sred[0] + 1e-16f);

    // pass 3: r = sum a[n] * x[n,:]  (per-wave register partials, LDS merge)
    float a0 = 0, a1 = 0, a2 = 0, a3 = 0, a4 = 0, a5 = 0, a6 = 0, a7 = 0;
    for (int n = s + wave; n < e; n += 8) {
        float aw = __expf(e_buf[n] - m) * inv;
        const float4* xp = (const float4*)(x + (size_t)n * FDIM + lane * 8);
        if (n + 8 < e)
            __builtin_prefetch(x + (size_t)(n + 8) * FDIM + lane * 8, 0, 0);
        float4 x0 = xp[0], x1 = xp[1];
        a0 += aw * x0.x; a1 += aw * x0.y; a2 += aw * x0.z; a3 += aw * x0.w;
        a4 += aw * x1.x; a5 += aw * x1.y; a6 += aw * x1.z; a7 += aw * x1.w;
    }
    int cb = lane * 8;
    atomicAdd(&racc[cb + 0], a0); atomicAdd(&racc[cb + 1], a1);
    atomicAdd(&racc[cb + 2], a2); atomicAdd(&racc[cb + 3], a3);
    atomicAdd(&racc[cb + 4], a4); atomicAdd(&racc[cb + 5], a5);
    atomicAdd(&racc[cb + 6], a6); atomicAdd(&racc[cb + 7], a7);
    __syncthreads();

    float rv = racc[tid];
    q_star[(size_t)g * (2 * FDIM) + FDIM + tid] = rv;     // r half of output
    Z[(size_t)g * KDIM + FDIM + tid] = (_Float16)rv;      // r part of next GEMM input
}

// ---------------- host side ----------------
extern "C" void kernel_launch(void* const* d_in, const int* in_sizes, int n_in,
                              void* d_out, int out_size, void* d_ws, size_t ws_size,
                              hipStream_t stream) {
    const float* x     = (const float*)d_in[0];
    const int*   batch = (const int*)d_in[1];
    // d_in[2] = size scalar (value recoverable from out_size)
    const float* Wih   = (const float*)d_in[3];
    const float* Whh   = (const float*)d_in[4];
    const float* bih   = (const float*)d_in[5];
    const float* bhh   = (const float*)d_in[6];

    int N = in_sizes[0] / FDIM;          // 250000
    int G = out_size / (2 * FDIM);       // 4096
    float* q_star = (float*)d_out;       // [G, 2F] written in place

    char* ws = (char*)d_ws;
    size_t off = 0;
    auto alloc = [&](size_t bytes) -> void* {
        void* p = ws + off;
        off = (off + bytes + 255) & ~(size_t)255;
        return p;
    };
    _Float16* Wf16  = (_Float16*)alloc((size_t)NGATE * KDIM * sizeof(_Float16));
    float*    bsum  = (float*)   alloc((size_t)NGATE * sizeof(float));
    _Float16* Z     = (_Float16*)alloc((size_t)G * KDIM * sizeof(_Float16));
    float*    gates = (float*)   alloc((size_t)G * NGATE * sizeof(float));
    float*    hbuf  = (float*)   alloc((size_t)G * FDIM * sizeof(float));
    float*    cbuf  = (float*)   alloc((size_t)G * FDIM * sizeof(float));
    float*    e_buf = (float*)   alloc((size_t)N * sizeof(float));
    int*      start = (int*)     alloc((size_t)(G + 1) * sizeof(int));
    (void)ws_size; (void)n_in;

    hipMemsetAsync(Z,    0, (size_t)G * KDIM * sizeof(_Float16), stream);
    hipMemsetAsync(cbuf, 0, (size_t)G * FDIM * sizeof(float), stream);

    prep_weights<<<(NGATE * KDIM + 255) / 256, 256, 0, stream>>>(Wih, Whh, Wf16);
    prep_bias<<<(NGATE + 255) / 256, 256, 0, stream>>>(bih, bhh, bsum);
    seg_start<<<(N + 255) / 256, 256, 0, stream>>>(batch, start, N, G);

    int gemm_waves  = ((G + 15) / 16) * (NGATE / 16);
    int gemm_blocks = (gemm_waves * 32 + 255) / 256;

    for (int t = 0; t < TSTEPS; ++t) {
        gemm_gates<<<gemm_blocks, 256, 0, stream>>>(Z, Wf16, gates, G);
        lstm_pointwise<<<(G * FDIM + 255) / 256, 256, 0, stream>>>(
            gates, bsum, hbuf, cbuf, q_star, Z, G);
        attn_fused<<<G, 256, 0, stream>>>(x, hbuf, start, e_buf, q_star, Z);
    }
}
